// ModulatedConv_7945689498059
// MI455X (gfx1250) — compile-verified, matrix operand order checked by reading, not run on hardware
//
#include <hip/hip_runtime.h>

// ---------------------------------------------------------------------------
// ModulatedConv2d (StyleGAN2), B=16, CIN=COUT=512, K=3, H=W=64, PAD=1.
// Implicit-GEMM on CDNA5 WMMA (bf16 in / f32 accumulate), wave32,
// double-buffered LDS fed by GLOBAL_LOAD_ASYNC_TO_LDS_B128 (ASYNCcnt path).
// ---------------------------------------------------------------------------

#define CIN   512
#define COUT  512
#define BATCH 16
#define HW    64
#define NPIX  4096
#define SCALE_F 0.014731391274719739f   // 1/sqrt(512*9)

// padded input layout: [b][66][72][CIN] bf16 (channel-last, zero borders)
#define PH 66
#define PW 72

typedef __bf16 bf16_t;
typedef __attribute__((ext_vector_type(16))) __bf16          v16bf;
typedef __attribute__((ext_vector_type(8)))  float           v8f;
typedef __attribute__((ext_vector_type(8)))  unsigned short  u16x8;
typedef __attribute__((ext_vector_type(16))) unsigned short  u16x16;

__device__ __forceinline__ unsigned short f32_to_bf16(float f) {
    unsigned int u = __builtin_bit_cast(unsigned int, f);
    u += 0x7FFFu + ((u >> 16) & 1u);            // round-to-nearest-even
    return (unsigned short)(u >> 16);
}

// -------------------------- stage 1: sum_k w^2 per (o,i) -------------------
__global__ void k_wsq(const float* __restrict__ weight, float* __restrict__ wsq) {
    int tid = blockIdx.x * 256 + threadIdx.x;          // o*512 + i
    if (tid >= COUT * CIN) return;
    const float* wp = weight + (size_t)tid * 9;
    float a = 0.f;
#pragma unroll
    for (int k = 0; k < 9; ++k) a += wp[k] * wp[k];
    wsq[tid] = a;
}

// -------------------------- stage 2: demod factor d[b,o] -------------------
__global__ void k_dinv(const float* __restrict__ s, const float* __restrict__ wsq,
                       float* __restrict__ dinv) {
    int tid = blockIdx.x * 256 + threadIdx.x;          // b*512 + o
    if (tid >= BATCH * COUT) return;
    int b = tid >> 9, o = tid & 511;
    const float* sp = s + b * CIN;
    const float* wp = wsq + (size_t)o * CIN;
    float a = 0.f;
    for (int i = 0; i < CIN; ++i) a += sp[i] * sp[i] * wp[i];
    dinv[tid] = rsqrtf(a * SCALE_F * SCALE_F + 1e-8f);
}

// ------------- stage 3: modulated bf16 weights, tap-major [b][tap][o][i] ---
__global__ void k_wmod(const float* __restrict__ weight, const float* __restrict__ s,
                       const float* __restrict__ dinv, unsigned short* __restrict__ wmod) {
    size_t tid = (size_t)blockIdx.x * 256 + threadIdx.x;
    if (tid >= (size_t)BATCH * 9 * COUT * CIN) return;
    int i   = (int)(tid & 511);
    int o   = (int)((tid >> 9) & 511);
    int tap = (int)((tid >> 18) % 9);
    int b   = (int)(tid / ((size_t)9 << 18));
    float v = weight[((size_t)o * CIN + i) * 9 + tap] * s[b * CIN + i]
            * SCALE_F * dinv[b * COUT + o];
    wmod[tid] = f32_to_bf16(v);
}

// ------------- stage 4: padded channel-last bf16 input ---------------------
__global__ void k_xpad(const float* __restrict__ x, unsigned short* __restrict__ xpad) {
    size_t tid = (size_t)blockIdx.x * 256 + threadIdx.x;   // ((b*66+h)*72+w)*512+i
    if (tid >= (size_t)BATCH * PH * PW * CIN) return;
    int i  = (int)(tid & 511);
    int wv = (int)((tid >> 9) % PW);
    int hv = (int)((tid / ((size_t)PW << 9)) % PH);
    int b  = (int)(tid / ((size_t)PH * PW << 9));
    float v = 0.f;
    if (hv >= 1 && hv <= HW && wv >= 1 && wv <= HW)
        v = x[(((size_t)b * CIN + i) * HW + (hv - 1)) * HW + (wv - 1)];
    xpad[tid] = f32_to_bf16(v);
}

// ------------- stage 5: implicit GEMM, async-LDS double buffer -------------
#define KPAD 40                       // 80B LDS pitch: conflict-free frag loads
#define AROWS 128                     // M tile
#define BROWS 256                     // N tile
#define BUFHW ((AROWS + BROWS) * KPAD)   // halfwords per buffer (15360)
#define KSTEPS 144                    // 9 taps * 16 chunks of 32

// one 16-byte async memory->LDS copy per lane (CDNA5 ASYNCcnt path)
__device__ __forceinline__ void async_cp16(unsigned lds_byte, const void* g) {
    asm volatile("global_load_async_to_lds_b128 %0, %1, off"
                 :: "v"(lds_byte), "v"((unsigned long long)(uintptr_t)g)
                 : "memory");
}
__device__ __forceinline__ void wait_async0() {
    asm volatile("s_wait_asynccnt 0x0" ::: "memory");
}

__device__ __forceinline__ v16bf make_frag(const unsigned short* row, int kb) {
    // ISA 16-bit operand layout: lane<16 -> K = kb..kb+7 and kb+16..kb+23
    union { u16x16 w; u16x8 h[2]; v16bf f; } u;
    u.h[0] = *(const u16x8*)(row + kb);
    u.h[1] = *(const u16x8*)(row + kb + 16);
    return u.f;
}

__global__ void __launch_bounds__(256)
k_conv(const unsigned short* __restrict__ wmod,
       const unsigned short* __restrict__ xpad,
       float* __restrict__ y) {
    __shared__ __attribute__((aligned(16))) unsigned short smem[2 * BUFHW];

    const int t    = threadIdx.x;
    const int b    = blockIdx.z;
    const int m0   = blockIdx.y * AROWS;         // COUT tile
    const int n0   = blockIdx.x * BROWS;         // pixel tile (4 image rows)
    const int wave = t >> 5;
    const int lane = t & 31;
    const int wm   = wave >> 2;                  // 0..1  (M direction)
    const int wn   = wave & 3;                   // 0..3  (N direction)
    const int lr   = lane & 15;
    const int kb   = (lane >> 4) * 8;

    const unsigned short* wmodB = wmod + (size_t)b * 9 * COUT * CIN;
    const unsigned short* xpadB = xpad + (size_t)b * PH * PW * CIN;
    const unsigned ldsbase = (unsigned)(uintptr_t)(void*)smem;

    // issue one k-step's tiles into LDS buffer `buf` via async copies
    auto stage = [&](int ks, int buf) {
        const int tap = ks >> 4, kc = ks & 15;
        const int kh = tap / 3, kw = tap - kh * 3;
        const unsigned short* wmodT = wmodB + ((size_t)tap * COUT + m0) * CIN + kc * 32;
        const unsigned bufbase = ldsbase + (unsigned)buf * (BUFHW * 2);
        // A: 128 rows x 32 k  -> 512 16B chunks, 2 per thread
#pragma unroll
        for (int r = 0; r < 2; ++r) {
            int c = t + r * 256;
            int row = c >> 2, q = c & 3;
            async_cp16(bufbase + (unsigned)(row * KPAD + q * 8) * 2,
                       wmodT + (size_t)row * CIN + q * 8);
        }
        // B: 256 pixels x 32 k -> 1024 16B chunks, 4 per thread
#pragma unroll
        for (int r = 0; r < 4; ++r) {
            int c = t + r * 256;
            int row = c >> 2, q = c & 3;
            int pg = n0 + row;
            int h  = (pg >> 6) + kh;
            int w  = (pg & 63) + kw;
            async_cp16(bufbase + (unsigned)((AROWS + row) * KPAD + q * 8) * 2,
                       xpadB + ((size_t)h * PW + w) * CIN + kc * 32 + q * 8);
        }
    };

    v8f acc[4][4] = {};                          // wave computes 64x64 of C

    stage(0, 0);                                 // prologue
#pragma unroll 2
    for (int ks = 0; ks < KSTEPS; ++ks) {
        const int cur = ks & 1;
        wait_async0();                           // own stage(ks) copies done
        __syncthreads();                         // all waves' copies done; prev reads done
        if (ks + 1 < KSTEPS) stage(ks + 1, cur ^ 1);   // overlap with WMMAs below

        const unsigned short* At = smem + cur * BUFHW;
        const unsigned short* Bt = At + AROWS * KPAD;
        v16bf af[4], bfv[4];
#pragma unroll
        for (int mi = 0; mi < 4; ++mi)
            af[mi] = make_frag(&At[(wm * 64 + mi * 16 + lr) * KPAD], kb);
#pragma unroll
        for (int ni = 0; ni < 4; ++ni)
            bfv[ni] = make_frag(&Bt[(wn * 64 + ni * 16 + lr) * KPAD], kb);
#pragma unroll
        for (int mi = 0; mi < 4; ++mi)
#pragma unroll
            for (int ni = 0; ni < 4; ++ni)
                acc[mi][ni] = __builtin_amdgcn_wmma_f32_16x16x32_bf16(
                    false, af[mi], false, bfv[ni],
                    (short)0, acc[mi][ni], false, false);
    }

    // ---- epilogue: D layout -> o = base+v+8*(lane>=16), p = base+lane%16 ----
    float* yB = y + (size_t)b * COUT * NPIX;
    const int mbase = m0 + wm * 64 + (lane >> 4) * 8;
    const int nbase = n0 + wn * 64 + lr;
#pragma unroll
    for (int mi = 0; mi < 4; ++mi)
#pragma unroll
        for (int ni = 0; ni < 4; ++ni)
#pragma unroll
            for (int v = 0; v < 8; ++v)
                yB[(size_t)(mbase + mi * 16 + v) * NPIX + (nbase + ni * 16)] =
                    acc[mi][ni][v];
}

// ---------------------------------------------------------------------------
extern "C" void kernel_launch(void* const* d_in, const int* in_sizes, int n_in,
                              void* d_out, int out_size, void* d_ws, size_t ws_size,
                              hipStream_t stream) {
    const float* x      = (const float*)d_in[0];   // [16,512,64,64]
    const float* s      = (const float*)d_in[1];   // [16,512]
    const float* weight = (const float*)d_in[2];   // [512,512,3,3]
    float* y = (float*)d_out;                      // [16,512,64,64]

    // workspace layout (~154.5 MB total)
    char* ws = (char*)d_ws;
    size_t off = 0;
    float* wsq  = (float*)(ws + off);  off += (size_t)COUT * CIN * 4;          // 1 MB
    float* dinv = (float*)(ws + off);  off += (size_t)BATCH * COUT * 4;        // 32 KB
    unsigned short* wmod = (unsigned short*)(ws + off);
    off += (size_t)BATCH * 9 * COUT * CIN * 2;                                  // 75.5 MB
    unsigned short* xpad = (unsigned short*)(ws + off);                         // 77.9 MB

    k_wsq <<<(COUT * CIN) / 256, 256, 0, stream>>>(weight, wsq);
    k_dinv<<<(BATCH * COUT) / 256, 256, 0, stream>>>(s, wsq, dinv);
    k_wmod<<<(unsigned)((size_t)BATCH * 9 * COUT * CIN / 256), 256, 0, stream>>>
          (weight, s, dinv, wmod);
    k_xpad<<<(unsigned)((size_t)BATCH * PH * PW * CIN / 256), 256, 0, stream>>>
          (x, xpad);

    dim3 grid(NPIX / BROWS, COUT / AROWS, BATCH);   // (16, 4, 16)
    k_conv<<<grid, 256, 0, stream>>>(wmod, xpad, y);
}
